// Autoencoder_65352222376356
// MI455X (gfx1250) — matrix operand order repaired and checked by reading, not compiled, not run on hardware
//
#include <hip/hip_runtime.h>
#include <stdint.h>

typedef unsigned short u16;

#define B_  128
#define T_  1024
#define F_  256
#define H_  512
#define NG_ 2048   /* 4H */
#define KE_ 768    /* F + H */

// LDS row strides in u16 elements: row bytes + 16B pad per 512B (TDM pad_enable)
#define LDE_ENC 792   /* (768*2 + 3*16)/2 */
#define LDE_DEC 528   /* (512*2 + 2*16)/2 */

typedef __attribute__((ext_vector_type(16))) __bf16 bf16x16;
typedef __attribute__((ext_vector_type(8)))  float  f32x8;
typedef __attribute__((ext_vector_type(4)))  unsigned int u32x4;
typedef __attribute__((ext_vector_type(8)))  int i32x8;
typedef __attribute__((ext_vector_type(4)))  int i32x4;

union FragU { uint4 q[2]; bf16x16 v; };

__device__ __forceinline__ u16 f2bf(float x) {
  union { float f; unsigned u; } a; a.f = x;
  unsigned r = a.u + 0x7FFFu + ((a.u >> 16) & 1u);
  return (u16)(r >> 16);
}
__device__ __forceinline__ float fast_sigmoid(float x) { return 1.f / (1.f + __expf(-x)); }
__device__ __forceinline__ float fast_tanh(float x)    { return 2.f / (1.f + __expf(-2.f * x)) - 1.f; }

__device__ __forceinline__ f32x8 zero8() {
  f32x8 z;
#pragma unroll
  for (int i = 0; i < 8; ++i) z[i] = 0.f;
  return z;
}

// A fragment: 16x32 bf16 tile from row-major [rows][ld] storage.
__device__ __forceinline__ bf16x16 ldfrag_a(const u16* base, int ld, int row0, int k0, int lane) {
  const int half = lane >> 4;
  const u16* p = base + (size_t)(row0 + (lane & 15)) * ld + k0 + half * 16;
  FragU f;
  f.q[0] = *(const uint4*)(p);
  f.q[1] = *(const uint4*)(p + 8);
  return f.v;
}

// B fragment: 32x16 (KxN) bf16 tile from N-major (transposed) [N][ld] global storage.
__device__ __forceinline__ bf16x16 ldfrag_bt(const u16* baseT, int ld, int n0, int k0, int lane) {
  const u16* p = baseT + (size_t)(n0 + (lane & 15)) * ld + k0 + (lane >> 4) * 16;
  FragU f;
  f.q[0] = *(const uint4*)(p);
  f.q[1] = *(const uint4*)(p + 8);
  return f.v;
}

// B fragment from LDS laid out by TDM with 16B pad per 512B: elem idx = row*lde + k + (k>>8)*8
__device__ __forceinline__ bf16x16 ldfrag_b_lds(const u16* lds, int lde, int n0, int k0, int lane) {
  const int row = n0 + (lane & 15);
  const int kb = k0 + (lane >> 4) * 16;
  const int i0 = row * lde + kb + ((kb >> 8) << 3);
  const int kb2 = kb + 8;
  const int i1 = row * lde + kb2 + ((kb2 >> 8) << 3);
  FragU f;
  f.q[0] = *(const uint4*)(lds + i0);
  f.q[1] = *(const uint4*)(lds + i1);
  return f.v;
}

__device__ __forceinline__ f32x8 wmma_bf16(bf16x16 a, bf16x16 b, f32x8 c) {
  return __builtin_amdgcn_wmma_f32_16x16x32_bf16(false, a, false, b, (short)0, c, false, false);
}

// Tensor Data Mover: load this block's weight slice (4 gates x 16 rows x rowBytes)
// as a 3D tile into LDS, with 4-dword padding after every 512 bytes.
__device__ __forceinline__ void tdm_load_weights(const u16* gsrc, unsigned ldsOff, int rowBytes) {
  const uint64_t ga = (uint64_t)(uintptr_t)gsrc;
  const int d0 = rowBytes >> 3;                 // row length in 8B units
  u32x4 g0;
  g0[0] = 1u;                                   // count=1, user mode, no gather
  g0[1] = ldsOff;                               // lds_addr (bytes)
  g0[2] = (unsigned)(ga & 0xFFFFFFFFu);         // global_addr[31:0]
  g0[3] = (unsigned)((ga >> 32) & 0x1FFFFFFu) | 0x80000000u;  // addr[56:32] | type=2
  i32x8 g1;
  g1[0] = (3 << 16) | (1 << 20) | (6 << 22) | (3 << 25); // data_size=8B, pad_en, 512B interval, 4dw pad
  g1[1] = (d0 & 0xFFFF) << 16;                  // tensor_dim0[15:0] @ bits63:48
  g1[2] = (2048 & 0xFFFF) << 16;                // dim0 hi=0 | tensor_dim1[15:0] @ bits95:80
  g1[3] = (d0 & 0xFFFF) << 16;                  // tensor_dim1 hi=0 | tile_dim0 @ bits127:112
  g1[4] = 16 | (4 << 16);                       // tile_dim1=16 rows/gate | tile_dim2=4 gates
  const long long s1 = (long long)H_ * d0;      // gate stride (rows*row) in 8B units
  g1[5] = d0;                                   // tensor_dim0_stride[31:0]
  g1[6] = (int)((s1 & 0xFFFFLL) << 16);         // stride0 hi16=0 | stride1[15:0]
  g1[7] = (int)(s1 >> 16);                      // stride1[47:16]
  i32x4 g2;
  g2[0] = 4;                                    // tensor_dim2 (gates)
  g2[1] = 0; g2[2] = 0; g2[3] = 0;
  i32x4 g3;
  g3[0] = 0; g3[1] = 0; g3[2] = 0; g3[3] = 0;
  i32x8 g4;                                     // extra group (clang-23 6-arg form): zero
#pragma unroll
  for (int i = 0; i < 8; ++i) g4[i] = 0;
  __builtin_amdgcn_tensor_load_to_lds(g0, g1, g2, g3, g4, 0);
}

// Device-scope split barrier among gridDim.x blocks sharing `cnt`.
__device__ __forceinline__ void grid_barrier(unsigned* cnt, unsigned target, int tid, int lane) {
  __syncthreads();
  if (tid == 0) {
    __threadfence();
    __hip_atomic_fetch_add(cnt, 1u, __ATOMIC_RELEASE, __HIP_MEMORY_SCOPE_AGENT);
  }
  if (lane == 0) {  // every wave acquires -> each CU's L0 sees fresh data
    while (__hip_atomic_load(cnt, __ATOMIC_ACQUIRE, __HIP_MEMORY_SCOPE_AGENT) < target)
      __builtin_amdgcn_s_sleep(1);
  }
  __syncthreads();
}

// ---------------- prep kernels ----------------

__global__ void k_init(unsigned* counters, u16* hF0, u16* hF1, u16* hB0, u16* hB1,
                       u16* hD0, u16* hD1) {
  int i = blockIdx.x * blockDim.x + threadIdx.x;
  if (i < 16) counters[i] = 0;
  int stride = gridDim.x * blockDim.x;
  for (int j = i; j < B_ * H_; j += stride) {
    hF0[j] = 0; hF1[j] = 0; hB0[j] = 0; hB1[j] = 0; hD0[j] = 0; hD1[j] = 0;
  }
}

__global__ void k_cvt(const float* __restrict__ src, u16* __restrict__ dst, size_t n) {
  size_t i = (size_t)blockIdx.x * blockDim.x + threadIdx.x;
  size_t stride = (size_t)gridDim.x * blockDim.x;
  for (; i < n; i += stride) dst[i] = f2bf(src[i]);
}

// dst[n*K + k] = bf16(src[k*N + n])
__global__ void k_transpose(const float* __restrict__ src, u16* __restrict__ dst, int K, int N) {
  size_t total = (size_t)K * N;
  size_t i = (size_t)blockIdx.x * blockDim.x + threadIdx.x;
  size_t stride = (size_t)gridDim.x * blockDim.x;
  for (; i < total; i += stride) {
    int n = (int)(i / K), k = (int)(i % K);
    dst[i] = f2bf(src[(size_t)k * N + n]);
  }
}

// dst[n*(KW+KU) + k] = bf16(k < KW ? W[k][n] : U[k-KW][n])
__global__ void k_concat_transpose(const float* __restrict__ W, const float* __restrict__ U,
                                   u16* __restrict__ dst, int KW, int KU, int N) {
  int K = KW + KU;
  size_t total = (size_t)K * N;
  size_t i = (size_t)blockIdx.x * blockDim.x + threadIdx.x;
  size_t stride = (size_t)gridDim.x * blockDim.x;
  for (; i < total; i += stride) {
    int n = (int)(i / K), k = (int)(i % K);
    float v = (k < KW) ? W[(size_t)k * N + n] : U[(size_t)(k - KW) * N + n];
    dst[i] = f2bf(v);
  }
}

// ---------------- encoder: persistent bidirectional LSTM ----------------

__global__ void __launch_bounds__(256)
enc_lstm(const u16* __restrict__ xbf,
         const u16* __restrict__ WcatF, const u16* __restrict__ WcatB,
         const float* __restrict__ biasF, const float* __restrict__ biasB,
         u16* __restrict__ hF0, u16* __restrict__ hF1,
         u16* __restrict__ hB0, u16* __restrict__ hB1,
         unsigned* __restrict__ counters)
{
  extern __shared__ u16 smem[];            // 64 rows x LDE_ENC
  const int dir = blockIdx.y;
  const int hcol0 = blockIdx.x * 16;
  const int tid = threadIdx.x;
  const int wave = tid >> 5, lane = tid & 31;
  const int m0 = wave * 16;
  const u16* Wt = dir ? WcatB : WcatF;
  const float* bias = dir ? biasB : biasF;
  u16* hb0 = dir ? hB0 : hF0;
  u16* hb1 = dir ? hB1 : hF1;
  unsigned* cnt = counters + dir;

  // TDM: stage weight slice (4 gates x 16 cols x 768 K, padded) into LDS once.
  if (tid < 32) {
    tdm_load_weights(Wt + (size_t)hcol0 * KE_, (unsigned)(uintptr_t)smem, KE_ * 2);
    __builtin_amdgcn_s_wait_tensorcnt(0);
  }
  __syncthreads();

  const int nloc = lane & 15;
  float bv[4];
#pragma unroll
  for (int g = 0; g < 4; ++g) bv[g] = bias[g * H_ + hcol0 + nloc];

  float cst[8];
#pragma unroll
  for (int r = 0; r < 8; ++r) cst[r] = 0.f;

  for (int step = 0; step < T_; ++step) {
    const int t = dir ? (T_ - 1 - step) : step;
    const u16* hprev = (step & 1) ? hb1 : hb0;
    u16* hnext = (step & 1) ? hb0 : hb1;

    f32x8 acc0 = zero8(), acc1 = zero8(), acc2 = zero8(), acc3 = zero8();
#pragma unroll 4
    for (int kt = 0; kt < KE_ / 32; ++kt) {
      const int k0 = kt * 32;
      bf16x16 a = (k0 < F_)
        ? ldfrag_a(xbf + (size_t)t * F_, T_ * F_, m0, k0, lane)       // x_t slice
        : ldfrag_a(hprev, H_, m0, k0 - F_, lane);                     // h_{t-1} slice
      bf16x16 b0 = ldfrag_b_lds(smem, LDE_ENC,  0, k0, lane);
      bf16x16 b1 = ldfrag_b_lds(smem, LDE_ENC, 16, k0, lane);
      bf16x16 b2 = ldfrag_b_lds(smem, LDE_ENC, 32, k0, lane);
      bf16x16 b3 = ldfrag_b_lds(smem, LDE_ENC, 48, k0, lane);
      acc0 = wmma_bf16(a, b0, acc0);
      acc1 = wmma_bf16(a, b1, acc1);
      acc2 = wmma_bf16(a, b2, acc2);
      acc3 = wmma_bf16(a, b3, acc3);
    }
#pragma unroll
    for (int r = 0; r < 8; ++r) {
      float iv = fast_sigmoid(acc0[r] + bv[0]);
      float fv = fast_sigmoid(acc1[r] + bv[1]);
      float gv = fast_tanh   (acc2[r] + bv[2]);
      float ov = fast_sigmoid(acc3[r] + bv[3]);
      float c  = fv * cst[r] + iv * gv;
      cst[r] = c;
      float h = ov * fast_tanh(c);
      int mrow = m0 + ((lane >> 4) << 3) + r;
      hnext[(size_t)mrow * H_ + hcol0 + nloc] = f2bf(h);
    }
    grid_barrier(cnt, (unsigned)(step + 1) * gridDim.x, tid, lane);
  }
  // final h lives in hF0 / hB0 (T even)
}

// ---------------- zd0 = latent @ Wd + bd  (once; RepeatVector folds away) ----------------

__global__ void __launch_bounds__(256)
zd0_gemm(const u16* __restrict__ hF, const u16* __restrict__ hB,
         const u16* __restrict__ WdT, const float* __restrict__ bd,
         float* __restrict__ zd0)
{
  const int wave = threadIdx.x >> 5, lane = threadIdx.x & 31;
  const int tile = blockIdx.x * 8 + wave;       // 8 x 128 = 1024 tiles
  const int mt = tile >> 7, nt = tile & 127;
  const int m0 = mt * 16, n0 = nt * 16;
  f32x8 acc = zero8();
  for (int kt = 0; kt < (2 * H_) / 32; ++kt) {
    const int k0 = kt * 32;
    bf16x16 a = (k0 < H_) ? ldfrag_a(hF, H_, m0, k0, lane)
                          : ldfrag_a(hB, H_, m0, k0 - H_, lane);
    bf16x16 b = ldfrag_bt(WdT, 2 * H_, n0, k0, lane);
    acc = wmma_bf16(a, b, acc);
  }
  const int nloc = lane & 15;
  const float bb = bd[n0 + nloc];
#pragma unroll
  for (int r = 0; r < 8; ++r) {
    int m = m0 + ((lane >> 4) << 3) + r;
    zd0[(size_t)m * NG_ + n0 + nloc] = acc[r] + bb;
  }
}

// ---------------- decoder: persistent LSTM, return_sequences ----------------

__global__ void __launch_bounds__(256)
dec_lstm(const float* __restrict__ zd0, const u16* __restrict__ UdT,
         u16* __restrict__ hD0, u16* __restrict__ hD1,
         u16* __restrict__ dech, unsigned* __restrict__ counters)
{
  extern __shared__ u16 smem[];            // 64 rows x LDE_DEC
  const int hcol0 = blockIdx.x * 16;
  const int tid = threadIdx.x;
  const int wave = tid >> 5, lane = tid & 31;
  const int m0 = wave * 16;
  unsigned* cnt = counters + 2;

  if (tid < 32) {
    tdm_load_weights(UdT + (size_t)hcol0 * H_, (unsigned)(uintptr_t)smem, H_ * 2);
    __builtin_amdgcn_s_wait_tensorcnt(0);
  }
  __syncthreads();

  const int nloc = lane & 15;

  // zd0 (= latent@Wd + bd) is time-invariant: hoist into registers.
  float z0[8], z1[8], z2[8], z3[8];
#pragma unroll
  for (int r = 0; r < 8; ++r) {
    int mrow = m0 + ((lane >> 4) << 3) + r;
    size_t base = (size_t)mrow * NG_ + hcol0 + nloc;
    z0[r] = zd0[base + 0 * H_];
    z1[r] = zd0[base + 1 * H_];
    z2[r] = zd0[base + 2 * H_];
    z3[r] = zd0[base + 3 * H_];
  }

  float cst[8];
#pragma unroll
  for (int r = 0; r < 8; ++r) cst[r] = 0.f;

  for (int step = 0; step < T_; ++step) {
    const u16* hprev = (step & 1) ? hD1 : hD0;
    u16* hnext = (step & 1) ? hD0 : hD1;

    f32x8 acc0 = zero8(), acc1 = zero8(), acc2 = zero8(), acc3 = zero8();
#pragma unroll 4
    for (int kt = 0; kt < H_ / 32; ++kt) {
      const int k0 = kt * 32;
      bf16x16 a  = ldfrag_a(hprev, H_, m0, k0, lane);
      bf16x16 b0 = ldfrag_b_lds(smem, LDE_DEC,  0, k0, lane);
      bf16x16 b1 = ldfrag_b_lds(smem, LDE_DEC, 16, k0, lane);
      bf16x16 b2 = ldfrag_b_lds(smem, LDE_DEC, 32, k0, lane);
      bf16x16 b3 = ldfrag_b_lds(smem, LDE_DEC, 48, k0, lane);
      acc0 = wmma_bf16(a, b0, acc0);
      acc1 = wmma_bf16(a, b1, acc1);
      acc2 = wmma_bf16(a, b2, acc2);
      acc3 = wmma_bf16(a, b3, acc3);
    }
#pragma unroll
    for (int r = 0; r < 8; ++r) {
      float iv = fast_sigmoid(acc0[r] + z0[r]);
      float fv = fast_sigmoid(acc1[r] + z1[r]);
      float gv = fast_tanh   (acc2[r] + z2[r]);
      float ov = fast_sigmoid(acc3[r] + z3[r]);
      float c  = fv * cst[r] + iv * gv;
      cst[r] = c;
      float h = ov * fast_tanh(c);
      int mrow = m0 + ((lane >> 4) << 3) + r;
      u16 hb = f2bf(h);
      hnext[(size_t)mrow * H_ + hcol0 + nloc] = hb;
      dech[((size_t)mrow * T_ + step) * H_ + hcol0 + nloc] = hb;   // return_sequences
    }
    grid_barrier(cnt, (unsigned)(step + 1) * gridDim.x, tid, lane);
  }
}

// ---------------- out = dec @ Wo + bo ----------------
// Each wave: one 16-row M tile x two 16-col N tiles (A fragment reused for 2 WMMAs).

__global__ void __launch_bounds__(256)
dense_out(const u16* __restrict__ dech, const u16* __restrict__ WoT,
          const float* __restrict__ bo, float* __restrict__ out)
{
  const int wave = threadIdx.x >> 5, lane = threadIdx.x & 31;
  const long job = (long)blockIdx.x * 8 + wave;   // 8192 m-tiles * 8 n-pairs
  const int mt = (int)(job >> 3), np = (int)(job & 7);
  const int m0 = mt * 16, n0 = np * 32;
  f32x8 accA = zero8(), accB = zero8();
#pragma unroll
  for (int kt = 0; kt < H_ / 32; ++kt) {
    const int k0 = kt * 32;
    bf16x16 a  = ldfrag_a(dech, H_, m0, k0, lane);
    bf16x16 bA = ldfrag_bt(WoT, H_, n0,      k0, lane);
    bf16x16 bB = ldfrag_bt(WoT, H_, n0 + 16, k0, lane);
    accA = wmma_bf16(a, bA, accA);
    accB = wmma_bf16(a, bB, accB);
  }
  const int nloc = lane & 15;
  const float bbA = bo[n0 + nloc];
  const float bbB = bo[n0 + 16 + nloc];
#pragma unroll
  for (int r = 0; r < 8; ++r) {
    int m = m0 + ((lane >> 4) << 3) + r;
    out[(size_t)m * F_ + n0 + nloc] = accA[r] + bbA;
    out[(size_t)m * F_ + n0 + 16 + nloc] = accB[r] + bbB;
  }
}

// ---------------- host ----------------

extern "C" void kernel_launch(void* const* d_in, const int* in_sizes, int n_in,
                              void* d_out, int out_size, void* d_ws, size_t ws_size,
                              hipStream_t stream) {
  (void)in_sizes; (void)n_in; (void)out_size; (void)ws_size;
  const float* x  = (const float*)d_in[0];
  const float* Wf = (const float*)d_in[1];
  const float* Uf = (const float*)d_in[2];
  const float* bf = (const float*)d_in[3];
  const float* Wb = (const float*)d_in[4];
  const float* Ub = (const float*)d_in[5];
  const float* bb = (const float*)d_in[6];
  const float* Wd = (const float*)d_in[7];
  const float* Ud = (const float*)d_in[8];
  const float* bd = (const float*)d_in[9];
  const float* Wo = (const float*)d_in[10];
  const float* bo = (const float*)d_in[11];
  float* out = (float*)d_out;

  char* ws = (char*)d_ws;
  size_t off = 0;
  auto take = [&](size_t bytes) -> char* {
    char* p = ws + off;
    off = (off + bytes + 255) & ~(size_t)255;
    return p;
  };
  unsigned* counters = (unsigned*)take(64);
  u16* hF0 = (u16*)take((size_t)B_ * H_ * 2);
  u16* hF1 = (u16*)take((size_t)B_ * H_ * 2);
  u16* hB0 = (u16*)take((size_t)B_ * H_ * 2);
  u16* hB1 = (u16*)take((size_t)B_ * H_ * 2);
  u16* hD0 = (u16*)take((size_t)B_ * H_ * 2);
  u16* hD1 = (u16*)take((size_t)B_ * H_ * 2);
  u16* xbf   = (u16*)take((size_t)B_ * T_ * F_ * 2);
  u16* WcatF = (u16*)take((size_t)NG_ * KE_ * 2);
  u16* WcatB = (u16*)take((size_t)NG_ * KE_ * 2);
  u16* UdT   = (u16*)take((size_t)NG_ * H_ * 2);
  u16* WdT   = (u16*)take((size_t)NG_ * 2 * H_ * 2);
  u16* WoT   = (u16*)take((size_t)F_ * H_ * 2);
  float* zd0 = (float*)take((size_t)B_ * NG_ * 4);
  u16* dech  = (u16*)take((size_t)B_ * T_ * H_ * 2);

  k_init<<<64, 256, 0, stream>>>(counters, hF0, hF1, hB0, hB1, hD0, hD1);
  k_cvt<<<8192, 256, 0, stream>>>(x, xbf, (size_t)B_ * T_ * F_);
  k_concat_transpose<<<2048, 256, 0, stream>>>(Wf, Uf, WcatF, F_, H_, NG_);
  k_concat_transpose<<<2048, 256, 0, stream>>>(Wb, Ub, WcatB, F_, H_, NG_);
  k_transpose<<<2048, 256, 0, stream>>>(Ud, UdT, H_, NG_);
  k_transpose<<<2048, 256, 0, stream>>>(Wd, WdT, 2 * H_, NG_);
  k_transpose<<<512, 256, 0, stream>>>(Wo, WoT, H_, F_);

  // persistent bidirectional encoder: 32 blocks/dir, TDM-staged weights in LDS
  enc_lstm<<<dim3(32, 2), 256, 64 * LDE_ENC * 2, stream>>>(
      xbf, WcatF, WcatB, bf, bb, hF0, hF1, hB0, hB1, counters);

  zd0_gemm<<<128, 256, 0, stream>>>(hF0, hB0, WdT, bd, zd0);

  dec_lstm<<<32, 256, 64 * LDE_DEC * 2, stream>>>(zd0, UdT, hD0, hD1, dech, counters);

  dense_out<<<8192, 256, 0, stream>>>(dech, WoT, bo, out);
}
